// XCY_44375602102956
// MI455X (gfx1250) — matrix operand order repaired and checked by reading, not compiled, not run on hardware
//
#include <hip/hip_runtime.h>

#define TKN 4096
#define CCH 256
#define BB  16
#define OCH 512
#define GG  1024   // b (dst) tokens per batch
#define NA  3072   // a (src) tokens per batch

typedef __attribute__((ext_vector_type(16))) __bf16 v16bf;
typedef __attribute__((ext_vector_type(8)))  float  v8f;

struct U4 { unsigned int x, y, z, w; };
struct U8 { U4 lo, hi; };

__device__ __forceinline__ unsigned short f2bf(float f) {
  unsigned int u = __builtin_bit_cast(unsigned int, f);
  u += 0x7FFFu + ((u >> 16) & 1u);           // round-to-nearest-even
  return (unsigned short)(u >> 16);
}

__device__ __forceinline__ float silu(float y) {
  return y / (1.0f + __expf(-y));
}

// ---- T: transpose x (B,C,T) -> xfT (B,T,C) so all later passes are coalesced
__global__ void k_transpose(const float* __restrict__ x, float* __restrict__ xfT) {
  __shared__ float tile[32][33];
  int b = blockIdx.y;
  int tTile = blockIdx.x >> 3;   // 0..127
  int cTile = blockIdx.x & 7;    // 0..7
  int lx = threadIdx.x & 31, ly = threadIdx.x >> 5;
  int c0 = cTile * 32, t0 = tTile * 32;
  const float* xb = x + (size_t)b * CCH * TKN;
  #pragma unroll
  for (int r = ly; r < 32; r += 8)
    tile[r][lx] = xb[(size_t)(c0 + r) * TKN + t0 + lx];
  __syncthreads();
  float* ob = xfT + (size_t)b * TKN * CCH;
  #pragma unroll
  for (int r = ly; r < 32; r += 8)
    ob[(size_t)(t0 + r) * CCH + c0 + lx] = tile[lx][r];
}

// ---- A: data-independent 3-NN table (replicates reference's s=55 / s=32 coords
//         and top_k tie-breaking: lower index first on equal distance)
__global__ void k_nn(int* __restrict__ nn) {
  int a = blockIdx.x * blockDim.x + threadIdx.x;
  if (a >= NA) return;
  int ta = (a / 3) * 4 + (a % 3);
  int ay = ta / 55, ax = ta % 55;          // int(sqrt(3072)) == 55
  int bd0 = 1 << 30, bd1 = 1 << 30, bd2 = 1 << 30;
  int bi0 = 0, bi1 = 0, bi2 = 0;
  for (int bi = 0; bi < GG; ++bi) {
    int tb = 4 * bi + 3;
    int by = tb >> 5, bx = tb & 31;        // int(sqrt(1024)) == 32
    int dy = ay - by, dx = ax - bx;
    int d = dy * dy + dx * dx;
    if (d < bd0)      { bd2=bd1; bi2=bi1; bd1=bd0; bi1=bi0; bd0=d; bi0=bi; }
    else if (d < bd1) { bd2=bd1; bi2=bi1; bd1=d;  bi1=bi; }
    else if (d < bd2) { bd2=d;   bi2=bi; }
  }
  nn[a*3+0] = bi0; nn[a*3+1] = bi1; nn[a*3+2] = bi2;
}

// ---- B: inverse L2 norm of each b token (wave per token)
__global__ void k_bnorm(const float* __restrict__ xfT, float* __restrict__ binv) {
  int wid  = blockIdx.x * 8 + (threadIdx.x >> 5);   // 0 .. BB*GG-1
  int lane = threadIdx.x & 31;
  int b  = wid >> 10;
  int gi = wid & 1023;
  const float* row = xfT + (size_t)(b * TKN + 4 * gi + 3) * CCH;
  float s = 0.f;
  for (int c = lane; c < CCH; c += 32) { float v = row[c]; s += v * v; }
  #pragma unroll
  for (int off = 16; off; off >>= 1) s += __shfl_xor(s, off, 32);
  if (lane == 0) binv[wid] = rsqrtf(s);
}

// ---- C: pick destination b-token per a-token (wave per a-token).
//         |a| is common across candidates -> only b inv-norms needed.
__global__ void k_assign(const float* __restrict__ xfT, const float* __restrict__ binv,
                         const int* __restrict__ nn, int* __restrict__ dsti) {
  int wid  = blockIdx.x * 8 + (threadIdx.x >> 5);   // 0 .. BB*NA-1
  int lane = threadIdx.x & 31;
  int b = wid / NA;
  int a = wid - b * NA;
  int ta = (a / 3) * 4 + (a % 3);
  int n0 = nn[a*3+0], n1 = nn[a*3+1], n2 = nn[a*3+2];
  const float* ra = xfT + (size_t)(b * TKN + ta) * CCH;
  const float* r0 = xfT + (size_t)(b * TKN + 4*n0 + 3) * CCH;
  const float* r1 = xfT + (size_t)(b * TKN + 4*n1 + 3) * CCH;
  const float* r2 = xfT + (size_t)(b * TKN + 4*n2 + 3) * CCH;
  float s0 = 0.f, s1 = 0.f, s2 = 0.f;
  for (int c = lane; c < CCH; c += 32) {
    float va = ra[c];
    s0 += va * r0[c]; s1 += va * r1[c]; s2 += va * r2[c];
  }
  #pragma unroll
  for (int off = 16; off; off >>= 1) {
    s0 += __shfl_xor(s0, off, 32);
    s1 += __shfl_xor(s1, off, 32);
    s2 += __shfl_xor(s2, off, 32);
  }
  if (lane == 0) {
    s0 *= binv[b*GG + n0]; s1 *= binv[b*GG + n1]; s2 *= binv[b*GG + n2];
    int best = n0; float bs = s0;                 // argmax, lowest x on tie
    if (s1 > bs) { bs = s1; best = n1; }
    if (s2 > bs) { bs = s2; best = n2; }
    dsti[wid] = best;
  }
}

// ---- D: init merged = dst own features (token-major), counts = 1
__global__ void k_minit(const float* __restrict__ xfT, float* __restrict__ merged,
                        float* __restrict__ counts) {
  int b = blockIdx.y, gi = blockIdx.x, c = threadIdx.x;
  merged[(size_t)(b * GG + gi) * CCH + c] =
      xfT[(size_t)(b * TKN + 4 * gi + 3) * CCH + c];
  if (c == 0) counts[b * GG + gi] = 1.0f;
}

// ---- E: scatter-add a tokens into their chosen b token
__global__ void k_scatter(const float* __restrict__ xfT, const int* __restrict__ dsti,
                          float* __restrict__ merged, float* __restrict__ counts) {
  int b = blockIdx.y;
  int c = threadIdx.x;
  int aBase = blockIdx.x * 256;
  for (int i = 0; i < 256; ++i) {
    int a = aBase + i;
    int d = dsti[b * NA + a];
    int ta = (a / 3) * 4 + (a % 3);
    atomicAdd(&merged[(size_t)(b * GG + d) * CCH + c],
              xfT[(size_t)(b * TKN + ta) * CCH + c]);
    if (c == 0) atomicAdd(&counts[b * GG + d], 1.0f);
  }
}

// ---- F: divide by counts, convert to bf16 token-major [b][t'][c]
__global__ void k_fin(const float* __restrict__ merged, const float* __restrict__ counts,
                      unsigned short* __restrict__ xmbf) {
  int b = blockIdx.y, gi = blockIdx.x, c = threadIdx.x;
  size_t row = (size_t)(b * GG + gi);
  float cnt = counts[row];
  xmbf[row * CCH + c] = f2bf(merged[row * CCH + c] / cnt);
}

// ---- prep: bf16 weights + folded BN scale/shift
__global__ void k_prep(const float* __restrict__ cw, const float* __restrict__ g,
                       const float* __restrict__ be, const float* __restrict__ mu,
                       const float* __restrict__ va, unsigned short* __restrict__ wbf,
                       float* __restrict__ invarr, float* __restrict__ b2) {
  int oc = blockIdx.x, c = threadIdx.x;
  wbf[oc * CCH + c] = f2bf(cw[oc * CCH + c]);
  if (c == 0) {
    float inv = g[oc] / sqrtf(va[oc] + 0.001f);
    invarr[oc] = inv;
    b2[oc] = be[oc] - mu[oc] * inv;
  }
}

// ---- G: WMMA GEMM. Wave computes 16(M) x 64(N) tile; K-loop of 8 bf16 WMMAs.
//  A (16x32 bf16): lane l<16 -> row m0+l, 16 consecutive K at k0;
//                  lane>=16  -> same row, k0+16.
//  B (32x16 bf16): lane n<16 -> col n0+n, 16 consecutive K at k0; lane>=16 -> k0+16.
//  C/D f32: VGPR r -> M=r (lanes 0-15) / M=r+8 (lanes 16-31), N = lane&15.
__global__ void __launch_bounds__(256) k_gemm(const unsigned short* __restrict__ wbf,
                       const unsigned short* __restrict__ xmbf,
                       const float* __restrict__ invarr, const float* __restrict__ b2,
                       float* __restrict__ out) {
  int wid  = blockIdx.x * 8 + (threadIdx.x >> 5);   // 0..8191
  int lane = threadIdx.x & 31;
  int b   = wid >> 9;          // 512 waves per batch
  int rem = wid & 511;
  int mt  = rem >> 4;          // 0..31
  int ng  = rem & 15;          // 0..15
  int m0  = mt << 4;
  int n0  = ng << 6;
  int sel = (lane >> 4) & 1;
  int l15 = lane & 15;

  v8f acc0 = {}, acc1 = {}, acc2 = {}, acc3 = {};

  const unsigned short* wrow = wbf + (size_t)(m0 + l15) * CCH + sel * 16;
  const unsigned short* xb   = xmbf + (size_t)b * GG * CCH;
  const unsigned short* bp0  = xb + (size_t)(n0 +  0 + l15) * CCH + sel * 16;
  const unsigned short* bp1  = xb + (size_t)(n0 + 16 + l15) * CCH + sel * 16;
  const unsigned short* bp2  = xb + (size_t)(n0 + 32 + l15) * CCH + sel * 16;
  const unsigned short* bp3  = xb + (size_t)(n0 + 48 + l15) * CCH + sel * 16;

  #pragma unroll
  for (int kk = 0; kk < 8; ++kk) {
    int k0 = kk * 32;
    U8 aw;
    aw.lo = *(const U4*)(wrow + k0);
    aw.hi = *(const U4*)(wrow + k0 + 8);
    v16bf av = __builtin_bit_cast(v16bf, aw);

    U8 bw0 = *(const U8*)(bp0 + k0);
    U8 bw1 = *(const U8*)(bp1 + k0);
    U8 bw2 = *(const U8*)(bp2 + k0);
    U8 bw3 = *(const U8*)(bp3 + k0);

    acc0 = __builtin_amdgcn_wmma_f32_16x16x32_bf16(false, av, false,
             __builtin_bit_cast(v16bf, bw0), (short)0, acc0, false, false);
    acc1 = __builtin_amdgcn_wmma_f32_16x16x32_bf16(false, av, false,
             __builtin_bit_cast(v16bf, bw1), (short)0, acc1, false, false);
    acc2 = __builtin_amdgcn_wmma_f32_16x16x32_bf16(false, av, false,
             __builtin_bit_cast(v16bf, bw2), (short)0, acc2, false, false);
    acc3 = __builtin_amdgcn_wmma_f32_16x16x32_bf16(false, av, false,
             __builtin_bit_cast(v16bf, bw3), (short)0, acc3, false, false);
  }

  float* ob = out + (size_t)b * OCH * GG;
  int colb = n0 + l15;
  #pragma unroll
  for (int r = 0; r < 8; ++r) {
    int oc = m0 + r + sel * 8;
    float inv = invarr[oc];
    float bt  = b2[oc];
    float* orow = ob + (size_t)oc * GG + colb;
    float y;
    y = acc0[r] * inv + bt; orow[0]  = silu(y);
    y = acc1[r] * inv + bt; orow[16] = silu(y);
    y = acc2[r] * inv + bt; orow[32] = silu(y);
    y = acc3[r] * inv + bt; orow[48] = silu(y);
  }
}

extern "C" void kernel_launch(void* const* d_in, const int* in_sizes, int n_in,
                              void* d_out, int out_size, void* d_ws, size_t ws_size,
                              hipStream_t stream) {
  const float* x  = (const float*)d_in[0];
  const float* cw = (const float*)d_in[1];
  const float* gg = (const float*)d_in[2];
  const float* be = (const float*)d_in[3];
  const float* mu = (const float*)d_in[4];
  const float* va = (const float*)d_in[5];
  float* out = (float*)d_out;

  char* w = (char*)d_ws;
  size_t off = 0;
  float* xfT   = (float*)(w + off);          off += (size_t)BB * TKN * CCH * 4;  // 64 MB
  float* merged= (float*)(w + off);          off += (size_t)BB * GG  * CCH * 4;  // 16 MB
  float* counts= (float*)(w + off);          off += (size_t)BB * GG * 4;
  unsigned short* xmbf = (unsigned short*)(w + off); off += (size_t)BB * GG * CCH * 2; // 8 MB
  unsigned short* wbf  = (unsigned short*)(w + off); off += (size_t)OCH * CCH * 2;
  float* invarr = (float*)(w + off);         off += OCH * 4;
  float* b2     = (float*)(w + off);         off += OCH * 4;
  int*   nn     = (int*)(w + off);           off += (size_t)NA * 3 * 4;
  int*   dsti   = (int*)(w + off);           off += (size_t)BB * NA * 4;
  float* binv   = (float*)(w + off);         off += (size_t)BB * GG * 4;

  k_transpose<<<dim3(128 * 8, BB), 256, 0, stream>>>(x, xfT);
  k_nn      <<<(NA + 255) / 256, 256, 0, stream>>>(nn);
  k_bnorm   <<<BB * GG / 8, 256, 0, stream>>>(xfT, binv);
  k_assign  <<<BB * NA / 8, 256, 0, stream>>>(xfT, binv, nn, dsti);
  k_minit   <<<dim3(GG, BB), 256, 0, stream>>>(xfT, merged, counts);
  k_scatter <<<dim3(NA / 256, BB), 256, 0, stream>>>(xfT, dsti, merged, counts);
  k_fin     <<<dim3(GG, BB), 256, 0, stream>>>(merged, counts, xmbf);
  k_prep    <<<OCH, 256, 0, stream>>>(cw, gg, be, mu, va, wbf, invarr, b2);
  k_gemm    <<<BB * 32 * 16 / 8, 256, 0, stream>>>(wbf, xmbf, invarr, b2, out);
}
